// GATNet_31679678775979
// MI455X (gfx1250) — matrix operand order repaired and checked by reading, not compiled, not run on hardware
//
#include <hip/hip_runtime.h>
#include <hip/hip_bf16.h>

// ============================================================================
// GATNet forward for MI455X (gfx1250), wave32 + WMMA bf16.
//
// FLOP budget: ~590 GFLOP in the JK-LSTM GEMMs, ~40 GFLOP conv GEMMs,
// ~10 GFLOP final linears -> all routed through v_wmma_f32_16x16x32_bf16.
// Memory budget: ~5.5 GB of edge gather traffic (~240us @ 23.3 TB/s) in the
// segment-softmax aggregation -> plain VMEM + float atomics.
// ============================================================================

typedef __attribute__((ext_vector_type(16))) __bf16 bf16x16;
typedef __attribute__((ext_vector_type(8)))  float  v8f;

#define HID  128
#define H2   192
#define G4   768   // 4*H2
#define NAA  26
#define EMB  123   // HID-5

// ---------------------------------------------------------------------------
// helpers
// ---------------------------------------------------------------------------
__device__ __forceinline__ unsigned short f2bf(float f) {
  union { float f; unsigned u; } x; x.f = f;
  unsigned r = (x.u + 0x7FFFu + ((x.u >> 16) & 1u)) >> 16;   // RNE
  return (unsigned short)r;
}

__device__ __forceinline__ float sigm(float x) { return 1.f / (1.f + __expf(-x)); }

__device__ __forceinline__ void atomicMaxF(float* addr, float val) {
  int* ia = (int*)addr;
  int old = __float_as_int(*addr);
  while (__int_as_float(old) < val) {
    int assumed = old;
    old = atomicCAS(ia, assumed, __float_as_int(val));
    if (old == assumed) break;
  }
}

// ---------------------------------------------------------------------------
// WMMA GEMM:  C[M,Nout] = A[M,K] * Bt[Nout,K]^T  (+ A2*B2t^T) (+ bias[col])
// A is f32 (converted to bf16 on the fly); Bt is pre-converted bf16 stored
// N-major so each lane's B fragment (K=32 slice) is ONE contiguous 32B load
// in contiguous k order k=kk+16*(l>>4)+s, same as A (lane n=l&15, k-pairs
// ascending, halves split by l>>4).
// A fragment layout (matches B): lane l row m=l&15,
// one contiguous 16-element k run at kk+g*16 (g=l>>4).
// One wave per block; wave computes a 16(M) x 128(N) tile (8 v8f accums).
// Requires: M%16==0, Nout%128==0, K%32==0. (Here M=100000=6250*16.)
// ---------------------------------------------------------------------------
__device__ __forceinline__ void wmma_k_loop(const float* __restrict__ A,
                                            const unsigned short* __restrict__ Btu,
                                            int K, int mrow, int lane, int col_base,
                                            v8f acc[8]) {
  const int g    = lane >> 4;
  const int nloc = lane & 15;
  const float* arow = A + (size_t)mrow * K;
  const __bf16* Bt  = reinterpret_cast<const __bf16*>(Btu);
  for (int kk = 0; kk < K; kk += 32) {
    __builtin_prefetch(arow + kk + 32, 0, 1);            // global_prefetch for next A chunk
    bf16x16 a;
    const float* p0 = arow + kk + g * 16;
    const float* p1 = p0 + 8;
#pragma unroll
    for (int j = 0; j < 8; ++j) { a[j] = (__bf16)p0[j]; a[8 + j] = (__bf16)p1[j]; }
#pragma unroll
    for (int t = 0; t < 8; ++t) {
      const bf16x16 b =
          *reinterpret_cast<const bf16x16*>(Bt + (size_t)(col_base + t * 16 + nloc) * K + kk + g * 16);
      acc[t] = __builtin_amdgcn_wmma_f32_16x16x32_bf16(false, a, false, b,
                                                       (short)0, acc[t], false, false);
    }
  }
}

__global__ __launch_bounds__(32)
void gemm_bf16_wmma(const float* __restrict__ A, const unsigned short* __restrict__ Bt, int K,
                    const float* __restrict__ A2, const unsigned short* __restrict__ B2t, int K2,
                    const float* __restrict__ bias, float* __restrict__ C, int Nout) {
  const int lane     = threadIdx.x;
  const int mrow     = blockIdx.x * 16 + (lane & 15);
  const int col_base = blockIdx.y * 128;
  v8f acc[8];
#pragma unroll
  for (int t = 0; t < 8; ++t)
#pragma unroll
    for (int r = 0; r < 8; ++r) acc[t][r] = 0.f;

  wmma_k_loop(A, Bt, K, mrow, lane, col_base, acc);
  if (A2) wmma_k_loop(A2, B2t, K2, mrow, lane, col_base, acc);   // uniform branch

  const int nloc  = lane & 15;
  const int mbase = blockIdx.x * 16 + ((lane >> 4) << 3);        // C/D layout: r -> M=r (+8 hi half)
#pragma unroll
  for (int t = 0; t < 8; ++t) {
    const int col = col_base + t * 16 + nloc;
    const float bv = bias ? bias[col] : 0.f;
#pragma unroll
    for (int r = 0; r < 8; ++r)
      C[(size_t)(mbase + r) * Nout + col] = acc[t][r] + bv;
  }
}

// ---------------------------------------------------------------------------
// weight prep
// ---------------------------------------------------------------------------
__global__ void k_convert_bf16(const float* __restrict__ s, unsigned short* __restrict__ d, size_t n) {
  size_t i = (size_t)blockIdx.x * blockDim.x + threadIdx.x;
  if (i < n) d[i] = f2bf(s[i]);
}
// src[K][Nn] -> dst[Nn][K] bf16
__global__ void k_transpose_bf16(const float* __restrict__ s, unsigned short* __restrict__ d, int K, int Nn) {
  int i = blockIdx.x * blockDim.x + threadIdx.x;
  if (i >= K * Nn) return;
  int n = i / K, k = i % K;
  d[i] = f2bf(s[(size_t)k * Nn + n]);
}
__global__ void k_add_vec(const float* __restrict__ a, const float* __restrict__ b,
                          float* __restrict__ o, int n) {
  int i = blockIdx.x * blockDim.x + threadIdx.x;
  if (i < n) o[i] = a[i] + b[i];
}
__global__ void k_fill(float* __restrict__ p, float v, size_t n) {
  size_t i = (size_t)blockIdx.x * blockDim.x + threadIdx.x;
  if (i < n) p[i] = v;
}

// ---------------------------------------------------------------------------
// node features: x0 = concat(emb_aa[aa] + emb_ss[ss], x_flt)
// ---------------------------------------------------------------------------
__global__ void k_init_x0(const int* __restrict__ xidx, const float* __restrict__ xflt,
                          const float* __restrict__ embaa, const float* __restrict__ embss,
                          float* __restrict__ x0, int N) {
  size_t i = (size_t)blockIdx.x * blockDim.x + threadIdx.x;
  if (i >= (size_t)N * HID) return;
  int n = (int)(i >> 7), c = (int)(i & 127);
  float v;
  if (c < EMB) {
    int aa = xidx[n * 2 + 0], ss = xidx[n * 2 + 1];
    v = embaa[(size_t)aa * EMB + c] + embss[(size_t)ss * EMB + c];
  } else v = xflt[(size_t)n * 5 + (c - EMB)];
  x0[i] = v;
}

__global__ void k_mean(const float* __restrict__ a, const float* __restrict__ b,
                       float* __restrict__ o, size_t n) {
  size_t i = (size_t)blockIdx.x * blockDim.x + threadIdx.x;
  if (i < n) o[i] = 0.5f * (a[i] + b[i]);
}

// ---------------------------------------------------------------------------
// GAT attention pieces
// ---------------------------------------------------------------------------
__global__ void k_att_scores(const float* __restrict__ xh, const float* __restrict__ att_src,
                             const float* __restrict__ att_dst, float* __restrict__ asrc,
                             float* __restrict__ adst, int N) {
  int i = blockIdx.x * blockDim.x + threadIdx.x;
  if (i >= N * 2) return;
  int n = i >> 1, h = i & 1;
  const float* row = xh + (size_t)n * 256 + h * 128;
  float s = 0.f, d = 0.f;
  for (int c = 0; c < 128; ++c) { s += row[c] * att_src[h * 128 + c]; d += row[c] * att_dst[h * 128 + c]; }
  asrc[i] = s; adst[i] = d;
}

__global__ void k_edge_pass1(const int* __restrict__ srcA, const int* __restrict__ dstA, int E, int Ep,
                             const float* __restrict__ asrc, const float* __restrict__ adst,
                             float* __restrict__ ev, float* __restrict__ mbuf) {
  int e = blockIdx.x * blockDim.x + threadIdx.x;
  if (e >= Ep) return;
  int s, d;
  if (e < E) { s = srcA[e]; d = dstA[e]; } else { s = d = e - E; }   // appended self-loops
#pragma unroll
  for (int h = 0; h < 2; ++h) {
    float v = asrc[s * 2 + h] + adst[d * 2 + h];
    v = v > 0.f ? v : 0.2f * v;                                      // leaky_relu(0.2)
    ev[(size_t)e * 2 + h] = v;
    atomicMaxF(&mbuf[d * 2 + h], v);
  }
}

__global__ void k_edge_pass2(const int* __restrict__ srcA, const int* __restrict__ dstA, int E, int Ep,
                             float* __restrict__ ev, const float* __restrict__ mbuf,
                             float* __restrict__ denom) {
  int e = blockIdx.x * blockDim.x + threadIdx.x;
  if (e >= Ep) return;
  int d = (e < E) ? dstA[e] : (e - E);
#pragma unroll
  for (int h = 0; h < 2; ++h) {
    float ex = __expf(ev[(size_t)e * 2 + h] - mbuf[d * 2 + h]);
    ev[(size_t)e * 2 + h] = ex;
    atomicAdd(&denom[d * 2 + h], ex);
  }
}

// wave per edge; lane handles 4 channels; averages the 2 heads (x0.5)
__global__ __launch_bounds__(256)
void k_edge_pass3(const int* __restrict__ srcA, const int* __restrict__ dstA, int E, int Ep,
                  const float* __restrict__ ev, const float* __restrict__ denom,
                  const float* __restrict__ xh, float* __restrict__ agg) {
  int w = (int)(((size_t)blockIdx.x * blockDim.x + threadIdx.x) >> 5);
  int lane = threadIdx.x & 31;
  if (w >= Ep) return;
  int s, d;
  if (w < E) { s = srcA[w]; d = dstA[w]; } else { s = d = w - E; }
  float a0 = ev[(size_t)w * 2 + 0] / (denom[d * 2 + 0] + 1e-16f);
  float a1 = ev[(size_t)w * 2 + 1] / (denom[d * 2 + 1] + 1e-16f);
  const float4 x0 = *(const float4*)(xh + (size_t)s * 256 + lane * 4);
  const float4 x1 = *(const float4*)(xh + (size_t)s * 256 + 128 + lane * 4);
  float* dp = agg + (size_t)d * 128 + lane * 4;
  atomicAdd(dp + 0, 0.5f * (x0.x * a0 + x1.x * a1));
  atomicAdd(dp + 1, 0.5f * (x0.y * a0 + x1.y * a1));
  atomicAdd(dp + 2, 0.5f * (x0.z * a0 + x1.z * a1));
  atomicAdd(dp + 3, 0.5f * (x0.w * a0 + x1.w * a1));
}

__global__ void k_conv_epilogue(const float* __restrict__ mean, const float* __restrict__ agg,
                                const float* __restrict__ bias, float* __restrict__ out, size_t n) {
  size_t i = (size_t)blockIdx.x * blockDim.x + threadIdx.x;
  if (i >= n) return;
  int c = (int)(i & 127);
  out[i] = mean[i] + fmaxf(agg[i] + bias[c], 0.f);
}

// ---------------------------------------------------------------------------
// LSTM cell (torch gate order i,f,g,o); one block per node, 192 threads
// ---------------------------------------------------------------------------
__global__ __launch_bounds__(H2)
void k_lstm_cell(const float* __restrict__ gates, float* __restrict__ h, float* __restrict__ c) {
  int n = blockIdx.x, j = threadIdx.x;
  const float* g = gates + (size_t)n * G4;
  size_t idx = (size_t)n * H2 + j;
  float gi = g[j], gf = g[H2 + j], gg = g[2 * H2 + j], go = g[3 * H2 + j];
  float cn = sigm(gf) * c[idx] + sigm(gi) * tanhf(gg);
  float hn = sigm(go) * tanhf(cn);
  c[idx] = cn; h[idx] = hn;
}

// alogit[t][n] += h[n,:] . attw   (wave per node, butterfly reduce)
__global__ __launch_bounds__(256)
void k_alogit(const float* __restrict__ h, const float* __restrict__ attw,
              float* __restrict__ alog_t, int N) {
  int w = (int)(((size_t)blockIdx.x * blockDim.x + threadIdx.x) >> 5);
  int lane = threadIdx.x & 31;
  if (w >= N) return;
  const float* row = h + (size_t)w * H2;
  float s = 0.f;
  for (int j = lane; j < H2; j += 32) s += row[j] * attw[j];
  for (int off = 16; off > 0; off >>= 1) s += __shfl_xor(s, off, 32);
  if (lane == 0) alog_t[w] += s;
}

// softmax over the 3 layer logits (jk_att_b constant cancels)
__global__ void k_softmax3(float* __restrict__ a, int N) {
  int n = blockIdx.x * blockDim.x + threadIdx.x;
  if (n >= N) return;
  float a0 = a[n], a1 = a[N + n], a2 = a[2 * N + n];
  float m = fmaxf(a0, fmaxf(a1, a2));
  float e0 = __expf(a0 - m), e1 = __expf(a1 - m), e2 = __expf(a2 - m);
  float inv = 1.f / (e0 + e1 + e2);
  a[n] = e0 * inv; a[N + n] = e1 * inv; a[2 * N + n] = e2 * inv;
}

__global__ void k_jk_combine(const float* __restrict__ xs_b, const float* __restrict__ w,
                             float* __restrict__ out, int N) {
  size_t i = (size_t)blockIdx.x * blockDim.x + threadIdx.x;
  size_t L = (size_t)N * HID;
  if (i >= L) return;
  int n = (int)(i >> 7);
  out[i] = xs_b[i] * w[n] + xs_b[L + i] * w[N + n] + xs_b[2 * L + i] * w[2 * N + n];
}

// final: alpha = softmax_i(outs_i . last_W); out = sum outs_i * alpha_i
__global__ __launch_bounds__(256)
void k_final_mix(const float* __restrict__ o0, const float* __restrict__ o1,
                 const float* __restrict__ o2, const float* __restrict__ o3,
                 const float* __restrict__ lastW, float* __restrict__ out, int N) {
  int w = (int)(((size_t)blockIdx.x * blockDim.x + threadIdx.x) >> 5);
  int lane = threadIdx.x & 31;
  if (w >= N) return;
  size_t base = (size_t)w * HID + lane * 4;
  float4 lw = *(const float4*)(lastW + lane * 4);
  float4 v0 = *(const float4*)(o0 + base);
  float4 v1 = *(const float4*)(o1 + base);
  float4 v2 = *(const float4*)(o2 + base);
  float4 v3 = *(const float4*)(o3 + base);
  float s0 = v0.x * lw.x + v0.y * lw.y + v0.z * lw.z + v0.w * lw.w;
  float s1 = v1.x * lw.x + v1.y * lw.y + v1.z * lw.z + v1.w * lw.w;
  float s2 = v2.x * lw.x + v2.y * lw.y + v2.z * lw.z + v2.w * lw.w;
  float s3 = v3.x * lw.x + v3.y * lw.y + v3.z * lw.z + v3.w * lw.w;
  for (int off = 16; off > 0; off >>= 1) {
    s0 += __shfl_xor(s0, off, 32); s1 += __shfl_xor(s1, off, 32);
    s2 += __shfl_xor(s2, off, 32); s3 += __shfl_xor(s3, off, 32);
  }
  float m = fmaxf(fmaxf(s0, s1), fmaxf(s2, s3));
  float e0 = __expf(s0 - m), e1 = __expf(s1 - m), e2 = __expf(s2 - m), e3 = __expf(s3 - m);
  float inv = 1.f / (e0 + e1 + e2 + e3);
  float a0 = e0 * inv, a1 = e1 * inv, a2 = e2 * inv, a3 = e3 * inv;
  float4 r;
  r.x = v0.x * a0 + v1.x * a1 + v2.x * a2 + v3.x * a3;
  r.y = v0.y * a0 + v1.y * a1 + v2.y * a2 + v3.y * a3;
  r.z = v0.z * a0 + v1.z * a1 + v2.z * a2 + v3.z * a3;
  r.w = v0.w * a0 + v1.w * a1 + v2.w * a2 + v3.w * a3;
  *(float4*)(out + base) = r;
}

// ===========================================================================
// host orchestration
// ===========================================================================
extern "C" void kernel_launch(void* const* d_in, const int* in_sizes, int n_in,
                              void* d_out, int out_size, void* d_ws, size_t ws_size,
                              hipStream_t stream) {
  const int N  = in_sizes[0] / 2;           // 100000
  const int E  = in_sizes[2] / 2;           // 800000
  const int Ep = E + N;                     // + self loops
  const int MT = N / 16;                    // 6250 wave row-tiles (N%16==0)

  const int*   x_idx   = (const int*)  d_in[0];
  const float* x_flt   = (const float*)d_in[1];
  const int*   eidx_l  = (const int*)  d_in[2];
  const int*   eidx_r  = (const int*)  d_in[3];
  const float* emb_aa  = (const float*)d_in[4];
  const float* emb_ss  = (const float*)d_in[5];
  const float* conv_W  = (const float*)d_in[6];
  const float* att_src = (const float*)d_in[7];
  const float* att_dst = (const float*)d_in[8];
  const float* conv_b  = (const float*)d_in[9];
  const float* lstm_Wih= (const float*)d_in[10];
  const float* lstm_Whh= (const float*)d_in[11];
  const float* lstm_bih= (const float*)d_in[12];
  const float* lstm_bhh= (const float*)d_in[13];
  const float* jk_attW = (const float*)d_in[14];
  // d_in[15] = jk_att_b: constant shift inside a softmax -> mathematically a no-op
  const float* dummy_W = (const float*)d_in[16];
  const float* dummy_b = (const float*)d_in[17];
  const float* fin_W   = (const float*)d_in[18];
  const float* fin_b   = (const float*)d_in[19];
  const float* last_W  = (const float*)d_in[20];
  float* outp = (float*)d_out;

  // ---- bump allocator over d_ws (~1.13 GB; union region time-sliced) -------
  size_t off = 0;
  char* W = (char*)d_ws;
  auto alloc = [&](size_t bytes) -> void* {
    void* p = W + off; off += (bytes + 255) & ~(size_t)255; return p;
  };
  const size_t NC = (size_t)N * HID;                 // node-feature elements
  float* xs   = (float*)alloc(6 * NC * 4);           // [branch][layer][N][128]
  float* x0   = (float*)alloc(2 * NC * 4);           // initial features
  float* mean = (float*)alloc(NC * 4);
  // union region: conv scratch | lstm scratch | final outs (largest = lstm)
  size_t uoff = off;
  float* xh    = (float*)alloc((size_t)N * 256 * 4);
  float* agg   = (float*)alloc(NC * 4);
  float* asrc  = (float*)alloc((size_t)N * 2 * 4);
  float* adst  = (float*)alloc((size_t)N * 2 * 4);
  float* mmax  = (float*)alloc((size_t)N * 2 * 4);
  float* denom = (float*)alloc((size_t)N * 2 * 4);
  float* evals = (float*)alloc((size_t)Ep * 2 * 4);
  size_t conv_end = off;
  off = uoff;
  float* gates  = (float*)alloc((size_t)N * G4 * 4);
  float* hbuf   = (float*)alloc((size_t)N * H2 * 4);
  float* cbuf   = (float*)alloc((size_t)N * H2 * 4);
  float* alogit = (float*)alloc((size_t)3 * N * 4);
  size_t lstm_end = off;
  off = uoff;
  float* outs = (float*)alloc(4 * NC * 4);
  size_t outs_end = off;
  off = conv_end > lstm_end ? conv_end : lstm_end;
  if (outs_end > off) off = outs_end;
  float* feats = (float*)alloc(4 * NC * 4);          // jk_l, sq_l, jk_r, sq_r (persist)
  unsigned short* convWt  = (unsigned short*)alloc((size_t)2 * 3 * 256 * 128 * 2);
  unsigned short* wihBf   = (unsigned short*)alloc((size_t)2 * 2 * G4 * HID * 2);
  unsigned short* whhBf   = (unsigned short*)alloc((size_t)2 * 2 * G4 * H2 * 2);
  unsigned short* dummyWt = (unsigned short*)alloc((size_t)2 * 128 * 128 * 2);
  unsigned short* finWt   = (unsigned short*)alloc((size_t)4 * 128 * 128 * 2);
  float* gbias = (float*)alloc((size_t)2 * 2 * G4 * 4);

  auto nblk = [](size_t n, int t) { return (unsigned)((n + t - 1) / t); };
  const int T = 256;

  // ---- weight prep (bf16, weights stored N-major for WMMA B fragments) ----
  for (int bl = 0; bl < 6; ++bl)
    k_transpose_bf16<<<nblk(128 * 256, T), T, 0, stream>>>(
        conv_W + (size_t)bl * 128 * 256, convWt + (size_t)bl * 256 * 128, 128, 256);
  k_convert_bf16<<<nblk((size_t)2 * 2 * G4 * HID, T), T, 0, stream>>>(
      lstm_Wih, wihBf, (size_t)2 * 2 * G4 * HID);            // x@Wih^T: Bt == Wih
  k_convert_bf16<<<nblk((size_t)2 * 2 * G4 * H2, T), T, 0, stream>>>(
      lstm_Whh, whhBf, (size_t)2 * 2 * G4 * H2);             // h@Whh^T: Bt == Whh
  for (int b = 0; b < 2; ++b)
    k_transpose_bf16<<<nblk(128 * 128, T), T, 0, stream>>>(
        dummy_W + (size_t)b * 128 * 128, dummyWt + (size_t)b * 128 * 128, 128, 128);
  for (int i = 0; i < 4; ++i)
    k_transpose_bf16<<<nblk(128 * 128, T), T, 0, stream>>>(
        fin_W + (size_t)i * 128 * 128, finWt + (size_t)i * 128 * 128, 128, 128);
  k_add_vec<<<nblk(2 * 2 * G4, T), T, 0, stream>>>(lstm_bih, lstm_bhh, gbias, 2 * 2 * G4);

  // ---- initial features ----------------------------------------------------
  for (int b = 0; b < 2; ++b)
    k_init_x0<<<nblk(NC, T), T, 0, stream>>>(
        x_idx, x_flt, emb_aa + (size_t)b * NAA * EMB, emb_ss + (size_t)b * 3 * EMB,
        x0 + (size_t)b * NC, N);

  // ---- 3 GAT conv layers, 2 branches --------------------------------------
  for (int L = 0; L < 3; ++L) {
    const float* xl = (L == 0) ? x0 : xs + (size_t)(0 * 3 + (L - 1)) * NC;
    const float* xr = (L == 0) ? x0 + NC : xs + (size_t)(1 * 3 + (L - 1)) * NC;
    k_mean<<<nblk(NC, T), T, 0, stream>>>(xl, xr, mean, NC);
    for (int b = 0; b < 2; ++b) {
      const float* xin = b ? xr : xl;
      const int* eidx  = b ? eidx_r : eidx_l;
      const int* srcA  = eidx;          // edge_index[0]
      const int* dstA  = eidx + E;      // edge_index[1]
      const size_t bl  = (size_t)b * 3 + L;
      gemm_bf16_wmma<<<dim3(MT, 2), 32, 0, stream>>>(
          xin, convWt + bl * 256 * 128, 128, nullptr, nullptr, 0, nullptr, xh, 256);
      k_att_scores<<<nblk((size_t)N * 2, T), T, 0, stream>>>(
          xh, att_src + bl * 2 * 128, att_dst + bl * 2 * 128, asrc, adst, N);
      k_fill<<<nblk((size_t)N * 2, T), T, 0, stream>>>(mmax, -1e30f, (size_t)N * 2);
      k_fill<<<nblk((size_t)N * 2, T), T, 0, stream>>>(denom, 0.f, (size_t)N * 2);
      k_fill<<<nblk(NC, T), T, 0, stream>>>(agg, 0.f, NC);
      k_edge_pass1<<<nblk(Ep, T), T, 0, stream>>>(srcA, dstA, E, Ep, asrc, adst, evals, mmax);
      k_edge_pass2<<<nblk(Ep, T), T, 0, stream>>>(srcA, dstA, E, Ep, evals, mmax, denom);
      k_edge_pass3<<<nblk((size_t)Ep * 32, T), T, 0, stream>>>(
          srcA, dstA, E, Ep, evals, denom, xh, agg);
      k_conv_epilogue<<<nblk(NC, T), T, 0, stream>>>(
          mean, agg, conv_b + bl * 128, xs + bl * NC, NC);
    }
  }

  // ---- JK bi-LSTM per branch (the FLOP heavyweight: 12 fused dual-GEMMs) --
  for (int b = 0; b < 2; ++b) {
    k_fill<<<nblk((size_t)3 * N, T), T, 0, stream>>>(alogit, 0.f, (size_t)3 * N);
    for (int d = 0; d < 2; ++d) {
      k_fill<<<nblk((size_t)N * H2, T), T, 0, stream>>>(hbuf, 0.f, (size_t)N * H2);
      k_fill<<<nblk((size_t)N * H2, T), T, 0, stream>>>(cbuf, 0.f, (size_t)N * H2);
      const size_t bd = (size_t)b * 2 + d;
      for (int s = 0; s < 3; ++s) {
        const int t = d ? (2 - s) : s;
        // gates = xs[t] @ Wih^T + h @ Whh^T + (bih+bhh), all in one WMMA kernel
        gemm_bf16_wmma<<<dim3(MT, 6), 32, 0, stream>>>(
            xs + ((size_t)b * 3 + t) * NC, wihBf + bd * G4 * HID, 128,
            hbuf, whhBf + bd * G4 * H2, 192,
            gbias + bd * G4, gates, G4);
        k_lstm_cell<<<N, H2, 0, stream>>>(gates, hbuf, cbuf);
        k_alogit<<<nblk((size_t)N * 32, T), T, 0, stream>>>(
            hbuf, jk_attW + (size_t)b * 2 * H2 + (size_t)d * H2, alogit + (size_t)t * N, N);
      }
    }
    k_softmax3<<<nblk(N, T), T, 0, stream>>>(alogit, N);
    k_jk_combine<<<nblk(NC, T), T, 0, stream>>>(
        xs + (size_t)b * 3 * NC, alogit, feats + (size_t)(b ? 2 : 0) * NC, N);
  }

  // ---- final linears + 4-way attention mix --------------------------------
  // feats = { jk_l, sq_l, jk_r, sq_r }
  gemm_bf16_wmma<<<dim3(MT, 1), 32, 0, stream>>>(
      feats + 0 * NC, dummyWt + 0, 128, nullptr, nullptr, 0, dummy_b + 0, feats + 1 * NC, 128);
  gemm_bf16_wmma<<<dim3(MT, 1), 32, 0, stream>>>(
      feats + 2 * NC, dummyWt + (size_t)128 * 128, 128, nullptr, nullptr, 0,
      dummy_b + 128, feats + 3 * NC, 128);
  for (int i = 0; i < 4; ++i)
    gemm_bf16_wmma<<<dim3(MT, 1), 32, 0, stream>>>(
        feats + (size_t)i * NC, finWt + (size_t)i * 128 * 128, 128,
        nullptr, nullptr, 0, fin_b + (size_t)i * 128, outs + (size_t)i * NC, 128);
  k_final_mix<<<nblk((size_t)N * 32, T), T, 0, stream>>>(
      outs + 0 * NC, outs + 1 * NC, outs + 2 * NC, outs + 3 * NC, last_W, outp, N);

  (void)n_in; (void)out_size; (void)ws_size;
}
